// LSTM_5471788335349
// MI455X (gfx1250) — matrix operand order repaired and checked by reading, not compiled, not run on hardware
//
#include <hip/hip_runtime.h>

// ---------------------------------------------------------------------------
// LSTM cell on MI455X (gfx1250), wave32 + v_wmma_f32_16x16x32_bf16.
//   gates[B,512] = [x | pre_h][B,256] @ Wt[256,512]   (bf16 in, f32 acc)
//   i,f,o = sigmoid ; c~ = tanh ; c = f*pre_c + i*c~ ; h = o*tanh(c)
// out = stack([h, c])  (float32, [2,B,128])
//
// M_BLK=128: each WG computes 128 rows x all 512 gate cols. 8 waves,
// wave = (wm in 0..1) x (wn in 0..3); each wave: 4 M-tiles x 8 N-tiles
// (4 gates x 2 u-tiles) -> every lane holds z_i,z_f,z_o,z_c for the same
// (row,u), so the LSTM elementwise is lane-local.
// ---------------------------------------------------------------------------

typedef __bf16 v4bf  __attribute__((ext_vector_type(4)));
typedef __bf16 v8bf  __attribute__((ext_vector_type(8)));
typedef __bf16 v16bf __attribute__((ext_vector_type(16)));
typedef float  v8f   __attribute__((ext_vector_type(8)));

#define B_TOTAL 131072
#define DCOLS   128      // input width D
#define UCOLS   128      // units U
#define KTOT    256      // D + U
#define NGATES  512      // 4*U
#define M_BLK   128      // rows per workgroup
#define LDA     264      // padded LDS row stride in bf16 elements (528 B)

// --- weight pre-pack: 8 f32 [128,128] matrices -> bf16 Wt[n=512][k=256] ----
// n = gate*128 + u ; k<128 -> W_g[k][u] ; k>=128 -> U_g[k-128][u]
// Column(n)-major over K so each WMMA B-fragment lane reads two contiguous
// 8-bf16 runs of its column, 16 k apart.
__global__ void lstm_pack_weights(const float* __restrict__ Wi,
                                  const float* __restrict__ Wf,
                                  const float* __restrict__ Wo,
                                  const float* __restrict__ Wc,
                                  const float* __restrict__ Ui,
                                  const float* __restrict__ Uf,
                                  const float* __restrict__ Uo,
                                  const float* __restrict__ Uc,
                                  __bf16* __restrict__ Wt) {
  int idx = blockIdx.x * blockDim.x + threadIdx.x;        // 512*256 total
  if (idx >= NGATES * KTOT) return;
  int n = idx / KTOT;
  int k = idx % KTOT;
  int g = n / UCOLS;
  int u = n % UCOLS;
  const float* Wg[4] = {Wi, Wf, Wo, Wc};
  const float* Ug[4] = {Ui, Uf, Uo, Uc};
  float v = (k < DCOLS) ? Wg[g][k * UCOLS + u]
                        : Ug[g][(k - DCOLS) * UCOLS + u];
  Wt[idx] = (__bf16)v;   // RNE
}

__device__ __forceinline__ float fast_sigmoid(float z) {
  return 1.0f / (1.0f + __expf(-z));
}
__device__ __forceinline__ float fast_tanh(float z) {
  float a = fabsf(z);
  float e = __expf(-2.0f * a);          // in (0,1], no overflow
  float t = (1.0f - e) / (1.0f + e);
  return copysignf(t, z);
}

__device__ __forceinline__ v16bf cat16(v8bf lo, v8bf hi) {
  return __builtin_shufflevector(lo, hi, 0,1,2,3,4,5,6,7,8,9,10,11,12,13,14,15);
}

// --- main fused kernel -----------------------------------------------------
__global__ __launch_bounds__(256)
void lstm_wmma(const float* __restrict__ pre_layer,   // [2,B,U]  (h then c)
               const float* __restrict__ x,           // [B,D]
               const __bf16* __restrict__ Wt,         // [512][256] bf16
               float* __restrict__ out) {             // [2,B,U]  (h then c)
  __shared__ __bf16 lds[M_BLK * LDA];                 // 67.6 KB

  const int tid     = threadIdx.x;
  const int rowBase = blockIdx.x * M_BLK;
  const float* pre_h = pre_layer;
  const float* pre_c = pre_layer + (size_t)B_TOTAL * UCOLS;

  // Stage [x | pre_h] rows rowBase..rowBase+127 into LDS as bf16.
  // Vectorized: each thread converts 4 consecutive floats per iteration.
  // 128*256 elems / (256 thr * 4) = 32 iters; chunks never straddle the
  // x|h boundary (k is a multiple of 4).
  for (int it = 0; it < (M_BLK * KTOT) / (256 * 4); ++it) {
    int flat = it * 1024 + tid * 4;
    int r = flat >> 8;            // row in tile
    int k = flat & 255;           // col in [0,256)
    float4 v;
    if (k < DCOLS) {
      v = *(const float4*)&x[(size_t)(rowBase + r) * DCOLS + k];
    } else {
      v = *(const float4*)&pre_h[(size_t)(rowBase + r) * UCOLS + (k - DCOLS)];
    }
    v4bf b = { (__bf16)v.x, (__bf16)v.y, (__bf16)v.z, (__bf16)v.w };
    *(v4bf*)&lds[r * LDA + k] = b;
  }
  __syncthreads();

  const int lane = tid & 31;
  const int w    = tid >> 5;
  const int wm   = w >> 2;          // 0..1 : M group (64 rows each)
  const int wn   = w & 3;           // 0..3 : u group (32 cols per gate)
  const int half = lane >> 4;       // 0/1
  const int l16  = lane & 15;

  // accumulators: [mi][g*2 + t], each a 16x16 f32 tile (8 VGPRs)
  v8f acc[4][8];
#pragma unroll
  for (int mi = 0; mi < 4; ++mi)
#pragma unroll
    for (int j = 0; j < 8; ++j) acc[mi][j] = (v8f){};

#pragma unroll
  for (int ks = 0; ks < 8; ++ks) {            // K = 8 * 32
    const int kb = ks * 32;

    // A fragments (ISA layout: lane holds row l16; K chunks at 8*half and 16+8*half)
    v16bf afrag[4];
#pragma unroll
    for (int mi = 0; mi < 4; ++mi) {
      const int rrow = 64 * wm + 16 * mi + l16;
      v8bf a0 = *(const v8bf*)&lds[rrow * LDA + kb + 8 * half];
      v8bf a1 = *(const v8bf*)&lds[rrow * LDA + kb + 16 + 8 * half];
      afrag[mi] = cat16(a0, a1);
    }

    // B fragments: lane holds column n, K chunks at kb+8*half and kb+16+8*half;
    // each B fragment feeds 4 M-tiles (0.75 loads per WMMA).
#pragma unroll
    for (int g = 0; g < 4; ++g) {
#pragma unroll
      for (int t = 0; t < 2; ++t) {
        const int n = g * 128 + 32 * wn + 16 * t + l16;
        const __bf16* bp = Wt + (size_t)n * KTOT + kb + 8 * half;
        v8bf b0 = *(const v8bf*)bp;
        v8bf b1 = *(const v8bf*)(bp + 16);
        v16bf bfrag = cat16(b0, b1);
        const int j = g * 2 + t;
#pragma unroll
        for (int mi = 0; mi < 4; ++mi) {
          acc[mi][j] = __builtin_amdgcn_wmma_f32_16x16x32_bf16(
              false, afrag[mi], false, bfrag, (short)0, acc[mi][j], false, false);
        }
      }
    }
  }

  // Elementwise LSTM. C/D layout: VGPR r -> row (r + 8*half), col l16.
#pragma unroll
  for (int mi = 0; mi < 4; ++mi) {
#pragma unroll
    for (int t = 0; t < 2; ++t) {
#pragma unroll
      for (int r = 0; r < 8; ++r) {
        const int m  = rowBase + 64 * wm + 16 * mi + 8 * half + r;
        const int uu = 32 * wn + 16 * t + l16;
        float zi = acc[mi][0 + t][r];
        float zf = acc[mi][2 + t][r];
        float zo = acc[mi][4 + t][r];
        float zc = acc[mi][6 + t][r];
        float it = fast_sigmoid(zi);
        float ft = fast_sigmoid(zf);
        float ot = fast_sigmoid(zo);
        float ct = fast_tanh(zc);
        float pc = pre_c[(size_t)m * UCOLS + uu];
        float c  = ft * pc + it * ct;
        float h  = ot * fast_tanh(c);
        out[(size_t)m * UCOLS + uu] = h;                               // h plane
        out[(size_t)B_TOTAL * UCOLS + (size_t)m * UCOLS + uu] = c;     // c plane
      }
    }
  }
}

// ---------------------------------------------------------------------------
extern "C" void kernel_launch(void* const* d_in, const int* in_sizes, int n_in,
                              void* d_out, int out_size, void* d_ws, size_t ws_size,
                              hipStream_t stream) {
  const float* pre_layer = (const float*)d_in[0];   // [2,B,U]
  const float* x         = (const float*)d_in[1];   // [B,D]
  const float* Wi = (const float*)d_in[2];
  const float* Wf = (const float*)d_in[3];
  const float* Wo = (const float*)d_in[4];
  const float* Wc = (const float*)d_in[5];
  const float* Ui = (const float*)d_in[6];
  const float* Uf = (const float*)d_in[7];
  const float* Uo = (const float*)d_in[8];
  const float* Uc = (const float*)d_in[9];

  __bf16* Wt = (__bf16*)d_ws;                       // needs 512*256*2 = 256 KB
  float*  out = (float*)d_out;

  lstm_pack_weights<<<(NGATES * KTOT + 255) / 256, 256, 0, stream>>>(
      Wi, Wf, Wo, Wc, Ui, Uf, Uo, Uc, Wt);

  lstm_wmma<<<B_TOTAL / M_BLK, 256, 0, stream>>>(pre_layer, x, Wt, out);
}
